// ChamfersLoss_6820408066664
// MI455X (gfx1250) — compile-verified, matrix operand order checked by reading, not run on hardware
//
#include <hip/hip_runtime.h>
#include <hip/hip_bf16.h>
#include <math.h>

typedef __attribute__((ext_vector_type(16))) __bf16 v16bf;
typedef __attribute__((ext_vector_type(8)))  float  v8f;

#define D_DIM 64
#define TILE 128
#define LSTR 72            // bf16 elems per LDS row: 144B stride, 16B aligned, conflict-free b128 reads
#define INF_BITS 0x7F800000u

// Build a 16x16x32-bf16 WMMA operand from an LDS row (row-major K).
// Element e of the 16-bit operand holds K = (e<8 ? e : e+8) + half*8 (+32*s),
// i.e. two contiguous 8-element runs -> two 16B LDS loads.
__device__ __forceinline__ v16bf load_frag_lds(const __bf16* rowp, int half, int s) {
  const __bf16* p0 = rowp + half * 8 + s * 32;
  const __bf16* p1 = p0 + 16;
  v16bf f;
#pragma unroll
  for (int i = 0; i < 8; ++i) { f[i] = p0[i]; f[i + 8] = p1[i]; }
  return f;
}

__global__ __launch_bounds__(256) void chamfer_pass1(
    const float* __restrict__ x, const float* __restrict__ y,
    float* __restrict__ rowmin, unsigned* __restrict__ colmin,
    int N, int M) {
  __shared__ __bf16 yhi[TILE * LSTR];
  __shared__ __bf16 ylo[TILE * LSTR];
  __shared__ float  x2s[TILE];
  __shared__ float  y2s[TILE];
  __shared__ float  y2p[256];
  __shared__ unsigned rowmin_s[TILE];
  __shared__ unsigned colmin_s[TILE];

  const int tid  = threadIdx.x;
  const int lane = tid & 31;
  const int wave = tid >> 5;
  const int half = lane >> 4;       // which 16-lane half of the wave
  const int l15  = lane & 15;
  const int Rbase = (wave >> 1) * 32;   // 4 row-groups x 2 col-groups of waves
  const int Cbase = (wave & 1) * 64;
  const int nbase = blockIdx.x * TILE;
  const int b     = blockIdx.y;

  // ---- A fragments (this block's 128 x-rows) held in registers for the whole
  // ---- M sweep; fp32 split into bf16 hi + residual lo.  Also |x|^2 per row.
  v16bf ahi[2][2], alo[2][2];
  {
    const float* xb = x + ((size_t)b * N + nbase) * D_DIM;
#pragma unroll
    for (int rt = 0; rt < 2; ++rt) {
      const float* xr = xb + (Rbase + rt * 16 + l15) * D_DIM;
      float p = 0.f;
#pragma unroll
      for (int s = 0; s < 2; ++s) {
        v16bf hi, lo;
#pragma unroll
        for (int e = 0; e < 16; ++e) {
          int k = ((e < 8) ? e : e + 8) + half * 8 + s * 32;
          float v = xr[k];
          __bf16 h = (__bf16)v;
          hi[e] = h;
          lo[e] = (__bf16)(v - (float)h);
          p = fmaf(v, v, p);
        }
        ahi[rt][s] = hi; alo[rt][s] = lo;
      }
      p += __shfl_xor(p, 16, 32);               // halves hold disjoint K subsets
      if (lane < 16) x2s[Rbase + rt * 16 + lane] = p;
    }
  }
  if (tid < TILE) rowmin_s[tid] = INF_BITS;
  __syncthreads();

  // |x|^2 indexed by C-tile element position (row = Rbase + rt*16 + half*8 + r)
  float x2v[2][8];
#pragma unroll
  for (int rt = 0; rt < 2; ++rt)
#pragma unroll
    for (int r = 0; r < 8; ++r)
      x2v[rt][r] = x2s[Rbase + rt * 16 + half * 8 + r];

  const float* yb = y + (size_t)b * M * D_DIM;
  const int col  = tid & 127;
  const int doff = (tid >> 7) * 32;

  for (int mbase = 0; mbase < M; mbase += TILE) {
    // ---- stage 128x64 y tile as bf16 hi/lo planes + |y|^2 partials ----
    {
      const float* yr = yb + (size_t)(mbase + col) * D_DIM + doff;
      __bf16* dhi = &yhi[col * LSTR + doff];
      __bf16* dlo = &ylo[col * LSTR + doff];
      float p = 0.f;
#pragma unroll
      for (int j = 0; j < 32; j += 4) {
        float4 q = *(const float4*)(yr + j);
        float qq[4] = {q.x, q.y, q.z, q.w};
#pragma unroll
        for (int c = 0; c < 4; ++c) {
          float v = qq[c];
          __bf16 h = (__bf16)v;
          dhi[j + c] = h;
          dlo[j + c] = (__bf16)(v - (float)h);
          p = fmaf(v, v, p);
        }
      }
      y2p[tid] = p;
      if (tid < TILE) colmin_s[tid] = INF_BITS;
    }
    // Prefetch next chunk's 128B region for this thread so the staging loads
    // after the compute phase hit WGP$/L2 (global_prefetch_b8; overlaps WMMA).
    if (mbase + TILE < M)
      __builtin_prefetch(yb + (size_t)(mbase + TILE + col) * D_DIM + doff, 0, 0);

    __syncthreads();
    if (tid < TILE) y2s[tid] = y2p[tid] + y2p[tid + TILE];
    __syncthreads();

    // ---- x.y^T for the 128x128 tile: bf16x3 emulation, fp32 accumulate ----
    v8f acc[2][4];
    const v8f vzero = {0.f, 0.f, 0.f, 0.f, 0.f, 0.f, 0.f, 0.f};
#pragma unroll
    for (int rt = 0; rt < 2; ++rt)
#pragma unroll
      for (int ct = 0; ct < 4; ++ct) acc[rt][ct] = vzero;

#pragma unroll
    for (int ct = 0; ct < 4; ++ct) {
      const __bf16* byh = &yhi[(Cbase + ct * 16 + l15) * LSTR];
      const __bf16* byl = &ylo[(Cbase + ct * 16 + l15) * LSTR];
#pragma unroll
      for (int s = 0; s < 2; ++s) {
        v16bf bh = load_frag_lds(byh, half, s);
        v16bf bl = load_frag_lds(byl, half, s);
#pragma unroll
        for (int rt = 0; rt < 2; ++rt) {
          acc[rt][ct] = __builtin_amdgcn_wmma_f32_16x16x32_bf16(
              false, ahi[rt][s], false, bh, (short)0, acc[rt][ct], false, false);
          acc[rt][ct] = __builtin_amdgcn_wmma_f32_16x16x32_bf16(
              false, ahi[rt][s], false, bl, (short)0, acc[rt][ct], false, false);
          acc[rt][ct] = __builtin_amdgcn_wmma_f32_16x16x32_bf16(
              false, alo[rt][s], false, bh, (short)0, acc[rt][ct], false, false);
        }
      }
    }

    // ---- fused epilogue: d2 = |x|^2 + |y|^2 - 2*xy; row/col mins ----
    // Clamp to 0 in-loop: keeps values non-negative so u32 bit ordering == f32
    // ordering for the atomicMin paths (and matches the reference's clamp).
    float rowm[2][8];
#pragma unroll
    for (int rt = 0; rt < 2; ++rt)
#pragma unroll
      for (int r = 0; r < 8; ++r) rowm[rt][r] = __builtin_inff();

#pragma unroll
    for (int ct = 0; ct < 4; ++ct) {
      float y2c = y2s[Cbase + ct * 16 + l15];
      float cm = __builtin_inff();
#pragma unroll
      for (int rt = 0; rt < 2; ++rt)
#pragma unroll
        for (int r = 0; r < 8; ++r) {
          float d = fmaf(-2.f, acc[rt][ct][r], x2v[rt][r] + y2c);
          d = fmaxf(d, 0.f);
          rowm[rt][r] = fminf(rowm[rt][r], d);
          cm = fminf(cm, d);
        }
      cm = fminf(cm, __shfl_xor(cm, 16, 32));   // combine the two row halves
      if (half == 0)
        atomicMin(&colmin_s[Cbase + ct * 16 + l15], __float_as_uint(cm));
    }
#pragma unroll
    for (int rt = 0; rt < 2; ++rt)
#pragma unroll
      for (int r = 0; r < 8; ++r) {
        float m = rowm[rt][r];
        m = fminf(m, __shfl_xor(m, 1, 32));
        m = fminf(m, __shfl_xor(m, 2, 32));
        m = fminf(m, __shfl_xor(m, 4, 32));
        m = fminf(m, __shfl_xor(m, 8, 32));
        if (l15 == 0)
          atomicMin(&rowmin_s[Rbase + rt * 16 + half * 8 + r], __float_as_uint(m));
      }
    __syncthreads();
    if (tid < TILE)
      atomicMin(&colmin[(size_t)b * M + mbase + tid], colmin_s[tid]);
    __syncthreads();
  }

  // rows are block-exclusive: plain store
  if (tid < TILE)
    rowmin[(size_t)b * N + nbase + tid] = __uint_as_float(rowmin_s[tid]);
}

__global__ void chamfer_init(unsigned* __restrict__ colmin, int n) {
  int i = blockIdx.x * 256 + threadIdx.x;
  if (i < n) colmin[i] = INF_BITS;
}

__global__ void chamfer_reduce(const float* __restrict__ rowmin,
                               const unsigned* __restrict__ colmin,
                               float* __restrict__ out,
                               int nrow, int ncol, float scale) {
  __shared__ float red[256];
  float s = 0.f;
  for (int i = threadIdx.x; i < nrow; i += 256) s += sqrtf(rowmin[i]);
  for (int i = threadIdx.x; i < ncol; i += 256) s += sqrtf(__uint_as_float(colmin[i]));
  red[threadIdx.x] = s;
  __syncthreads();
  for (int off = 128; off > 0; off >>= 1) {
    if (threadIdx.x < off) red[threadIdx.x] += red[threadIdx.x + off];
    __syncthreads();
  }
  if (threadIdx.x == 0) out[0] = red[0] * scale;
}

extern "C" void kernel_launch(void* const* d_in, const int* in_sizes, int n_in,
                              void* d_out, int out_size, void* d_ws, size_t ws_size,
                              hipStream_t stream) {
  const float* x = (const float*)d_in[0];
  const float* y = (const float*)d_in[1];
  const int Bb = 8;                       // per reference setup
  const int N = in_sizes[0] / (Bb * D_DIM);
  const int M = in_sizes[1] / (Bb * D_DIM);

  float*    rowmin = (float*)d_ws;                                    // B*N f32
  unsigned* colmin = (unsigned*)((char*)d_ws + (size_t)Bb * N * 4);   // B*M u32

  chamfer_init<<<(Bb * M + 255) / 256, 256, 0, stream>>>(colmin, Bb * M);

  dim3 grid(N / TILE, Bb);
  chamfer_pass1<<<grid, 256, 0, stream>>>(x, y, rowmin, colmin, N, M);

  chamfer_reduce<<<1, 256, 0, stream>>>(rowmin, colmin, (float*)d_out,
                                        Bb * N, Bb * M,
                                        1.0f / ((float)Bb * (float)M));
}
